// LSTMTS_79800492360026
// MI455X (gfx1250) — matrix operand order repaired and checked by reading, not compiled, not run on hardware
//
#include <hip/hip_runtime.h>
#include <hip/hip_bf16.h>
#include <math.h>

// ---------------------------------------------------------------------------
// LSTM over SEQ=8192 timesteps, D_IN=256, H=2048, D_OUT=64 on MI455X (gfx1250)
//   Phase 1: xg = x @ W_ih^T + (b_ih+b_hh)    -> bf16 WMMA GEMM (f32 accum)
//   Phase 2: sequential recurrence            -> persistent kernel, VALU FMAs,
//            W_hh bf16 (32MB, L2-resident), h staged in LDS, c in registers,
//            grid-wide barrier per step
//   Phase 3: out = sigmoid(hs @ W_fc^T + b_fc)-> bf16 WMMA GEMM + sigmoid
// ---------------------------------------------------------------------------

#define SEQ   8192
#define D_IN  256
#define H     2048
#define G4    (4 * H)      // 8192 gate rows
#define D_OUT 64
#define NB    64           // persistent blocks (co-resident; 1 per WGP max load)

typedef __attribute__((ext_vector_type(16))) __bf16 v16bf;
typedef __attribute__((ext_vector_type(8)))  float  v8f;

struct __attribute__((aligned(16))) U4 { unsigned x, y, z, w; };
union  BfFrag { v16bf v; U4 q[2]; };

// ---------------- helpers ----------------
__device__ __forceinline__ unsigned short f2bf(float f) {
    unsigned u = __float_as_uint(f);
    u += 0x7FFFu + ((u >> 16) & 1u);          // round-to-nearest-even
    return (unsigned short)(u >> 16);
}
__device__ __forceinline__ float bflo(unsigned x) { return __uint_as_float(x << 16); }
__device__ __forceinline__ float bfhi(unsigned x) { return __uint_as_float(x & 0xffff0000u); }
__device__ __forceinline__ float sigmoidf(float z) { return 1.0f / (1.0f + __expf(-z)); }

// A fragment: 16x32 bf16, row-major source (lda = K-stride in elements).
// Lane L: row M=L%16; K = koff + {0..7} in v[0..7], koff+16+{0..7} in v[8..15],
// koff = (L<16 ? 0 : 8).   (ISA 7.12.2, 16-bit A 16x32)
__device__ __forceinline__ v16bf load_a_frag(const unsigned short* A, int lda,
                                             int row0, int k0, int lane) {
    int m    = lane & 15;
    int koff = (lane & 16) ? 8 : 0;
    const unsigned short* p = A + (size_t)(row0 + m) * lda + k0 + koff;
    BfFrag f;
    f.q[0] = *(const U4*)(p);
    f.q[1] = *(const U4*)(p + 16);
    return f.v;
}

// B fragment: 32x16 bf16 where B[k,n] = W[n,k] (W row-major, ldw = K-stride).
// Lane L: column N=L%16; K = kb + {0..7} in v[0..7], kb+16+{0..7} in v[8..15], kb=(L<16?0:8).
__device__ __forceinline__ v16bf load_b_frag(const unsigned short* W, int ldw,
                                             int n0, int k0, int lane) {
    int n  = lane & 15;
    int kb = (lane & 16) ? 8 : 0;
    const unsigned short* p = W + (size_t)(n0 + n) * ldw + k0 + kb;
    BfFrag f;
    f.q[0] = *(const U4*)(p);
    f.q[1] = *(const U4*)(p + 16);
    return f.v;
}

// ---------------- prep kernels ----------------
__global__ void k_f2bf(const float* __restrict__ s, unsigned short* __restrict__ d, int n) {
    int i = blockIdx.x * blockDim.x + threadIdx.x;
    if (i < n) d[i] = f2bf(s[i]);
}

__global__ void k_init(const float* __restrict__ b_ih, const float* __restrict__ b_hh,
                       float* __restrict__ bsum, const float* __restrict__ hprev,
                       float* __restrict__ g_h, unsigned* __restrict__ bar) {
    int i = blockIdx.x * blockDim.x + threadIdx.x;
    if (i < G4)  bsum[i] = b_ih[i] + b_hh[i];
    if (i < H)   g_h[i]  = hprev[i];   // double-buffer slot 0
    if (i < 2)   bar[i]  = 0u;         // arrive count, generation
}

// ---------------- Phase 1: xg GEMM (WMMA) ----------------
// 512x512 tiles of 16x16; 8 waves/block, one tile per wave; K = 256.
__global__ void k_gemm_xg(const unsigned short* __restrict__ xb,
                          const unsigned short* __restrict__ wib,
                          const float* __restrict__ bsum,
                          float* __restrict__ xg) {
    int wave = threadIdx.x >> 5;
    int lane = threadIdx.x & 31;
    int tile = blockIdx.x * 8 + wave;       // [0, 262144)
    int tm = tile >> 9, tn = tile & 511;
    int t0 = tm * 16,  r0 = tn * 16;

    v8f acc = {};
#pragma unroll
    for (int k0 = 0; k0 < D_IN; k0 += 32) {
        v16bf a = load_a_frag(xb,  D_IN, t0, k0, lane);
        v16bf b = load_b_frag(wib, D_IN, r0, k0, lane);
        acc = __builtin_amdgcn_wmma_f32_16x16x32_bf16(false, a, false, b,
                                                      (short)0, acc, false, false);
    }
    int n  = lane & 15;
    int mb = (lane & 16) ? 8 : 0;
    float bias = bsum[r0 + n];
#pragma unroll
    for (int v = 0; v < 8; ++v)
        xg[(size_t)(t0 + mb + v) * G4 + r0 + n] = acc[v] + bias;
}

// ---------------- Phase 2: persistent recurrence ----------------
// Block b owns hidden units j in [32b, 32b+32) and their 4 gate rows
// (rows g*2048 + 32b + u, g=0..3).  256 threads: 2 threads per gate row,
// each covering half of K=2048 against LDS-resident h.
__global__ void k_lstm(const unsigned short* __restrict__ whh,
                       const float* __restrict__ xg,
                       const float* __restrict__ cprev,
                       float* __restrict__ g_h,          // double buffer [2][H]
                       unsigned short* __restrict__ hs,  // [SEQ][H] bf16
                       float* __restrict__ d_out,        // tail: hT, cT
                       unsigned* __restrict__ bar) {
    __shared__ float sh_h[H];
    __shared__ float sh_part[256];
    __shared__ float sh_z[128];

    const int tid  = threadIdx.x;
    const int b    = blockIdx.x;
    const int rt   = tid >> 1;          // 0..127 gate-row task
    const int kh   = tid & 1;           // K half
    const int gate = rt >> 5;           // 0..3 (i,f,g,o)
    const int u    = rt & 31;           // unit within block
    const int row  = gate * H + b * 32 + u;
    const unsigned short* wrow = whh + (size_t)row * H + kh * 1024;

    float creg = (tid < 32) ? cprev[b * 32 + tid] : 0.0f;

    for (int t = 0; t < SEQ; ++t) {
        const float* hin  = g_h + (size_t)(t & 1) * H;
        float*       hout = g_h + (size_t)((t + 1) & 1) * H;

        // stage h into LDS
        for (int i = tid; i < H; i += 256) sh_h[i] = hin[i];
        __syncthreads();

        // half-row dot product vs LDS h (bf16 weights, 16B loads)
        float acc = 0.0f;
        const float* hp = sh_h + kh * 1024;
#pragma unroll 2
        for (int k = 0; k < 1024; k += 8) {
            U4 q = *(const U4*)(wrow + k);
            acc = fmaf(bflo(q.x), hp[k + 0], acc);
            acc = fmaf(bfhi(q.x), hp[k + 1], acc);
            acc = fmaf(bflo(q.y), hp[k + 2], acc);
            acc = fmaf(bfhi(q.y), hp[k + 3], acc);
            acc = fmaf(bflo(q.z), hp[k + 4], acc);
            acc = fmaf(bfhi(q.z), hp[k + 5], acc);
            acc = fmaf(bflo(q.w), hp[k + 6], acc);
            acc = fmaf(bfhi(q.w), hp[k + 7], acc);
        }
        sh_part[tid] = acc;
        __syncthreads();

        if (tid < 128) {
            int g2 = tid >> 5, u2 = tid & 31;
            float z = sh_part[2 * tid] + sh_part[2 * tid + 1]
                    + xg[(size_t)t * G4 + g2 * H + b * 32 + u2];
            sh_z[tid] = z;
        }
        __syncthreads();

        if (tid < 32) {
            float zi = sh_z[tid],      zf = sh_z[32 + tid];
            float zg = sh_z[64 + tid], zo = sh_z[96 + tid];
            float ig = sigmoidf(zi), fg = sigmoidf(zf);
            float gg = tanhf(zg),    og = sigmoidf(zo);
            creg = fg * creg + ig * gg;
            float hn = og * tanhf(creg);
            int j = b * 32 + tid;
            hout[j] = hn;
            hs[(size_t)t * H + j] = f2bf(hn);
            if (t == SEQ - 1) {
                d_out[(size_t)SEQ * D_OUT + j]     = hn;    // hT
                d_out[(size_t)SEQ * D_OUT + H + j] = creg;  // cT
            }
        }
        __threadfence();

        // grid-wide barrier: arrive count + monotonically increasing generation
        __syncthreads();
        if (tid == 0) {
            unsigned arrived = atomicAdd(&bar[0], 1u);
            if (arrived == NB - 1) {
                bar[0] = 0u;
                __threadfence();
                atomicAdd(&bar[1], 1u);
            } else {
                unsigned target = (unsigned)(t + 1);
                while (*(volatile unsigned*)&bar[1] < target)
                    __builtin_amdgcn_s_sleep(2);
            }
            __threadfence();
        }
        __syncthreads();
    }
}

// ---------------- Phase 3: FC GEMM (WMMA) + sigmoid ----------------
// 512x4 tiles of 16x16; 8 waves/block -> 256 blocks; K = 2048.
__global__ void k_gemm_fc(const unsigned short* __restrict__ hsb,
                          const unsigned short* __restrict__ wfb,
                          const float* __restrict__ b_fc,
                          float* __restrict__ out) {
    int wave = threadIdx.x >> 5;
    int lane = threadIdx.x & 31;
    int tile = blockIdx.x * 8 + wave;      // [0, 2048)
    int tm = tile >> 2, tn = tile & 3;
    int t0 = tm * 16,  n0 = tn * 16;

    v8f acc = {};
#pragma unroll 4
    for (int k0 = 0; k0 < H; k0 += 32) {
        v16bf a = load_a_frag(hsb, H, t0, k0, lane);
        v16bf b = load_b_frag(wfb, H, n0, k0, lane);
        acc = __builtin_amdgcn_wmma_f32_16x16x32_bf16(false, a, false, b,
                                                      (short)0, acc, false, false);
    }
    int n  = lane & 15;
    int mb = (lane & 16) ? 8 : 0;
    float bias = b_fc[n0 + n];
#pragma unroll
    for (int v = 0; v < 8; ++v)
        out[(size_t)(t0 + mb + v) * D_OUT + n0 + n] = sigmoidf(acc[v] + bias);
}

// ---------------- host ----------------
extern "C" void kernel_launch(void* const* d_in, const int* in_sizes, int n_in,
                              void* d_out, int out_size, void* d_ws, size_t ws_size,
                              hipStream_t stream) {
    (void)in_sizes; (void)n_in; (void)out_size; (void)ws_size;

    const float* x     = (const float*)d_in[0];   // [SEQ, D_IN]
    const float* hprev = (const float*)d_in[1];   // [1,1,H]
    const float* cprev = (const float*)d_in[2];   // [1,1,H]
    const float* W_ih  = (const float*)d_in[3];   // [4H, D_IN]
    const float* W_hh  = (const float*)d_in[4];   // [4H, H]
    const float* b_ih  = (const float*)d_in[5];   // [4H]
    const float* b_hh  = (const float*)d_in[6];   // [4H]
    const float* W_fc  = (const float*)d_in[7];   // [D_OUT, H]
    const float* b_fc  = (const float*)d_in[8];   // [D_OUT]
    float* out = (float*)d_out;                   // out | hT | cT

    // workspace layout (bytes, 256-aligned)
    char* ws = (char*)d_ws;
    float*          xg   = (float*)(ws);                                  // 256 MB
    unsigned short* whhb = (unsigned short*)(ws + 268435456);             //  32 MB
    unsigned short* hsb  = (unsigned short*)(ws + 268435456 + 33554432);  //  32 MB
    unsigned short* xb   = (unsigned short*)(ws + 268435456 + 67108864);  //   4 MB
    unsigned short* wib  = (unsigned short*)(ws + 268435456 + 71303168);  //   4 MB
    unsigned short* wfb  = (unsigned short*)(ws + 268435456 + 75497472);  // 256 KB
    float*          bsum = (float*)(ws + 268435456 + 75759616);          //  32 KB
    float*          g_h  = (float*)(ws + 268435456 + 75792384);          //  16 KB (2xH)
    unsigned*       bar  = (unsigned*)(ws + 268435456 + 75808768);

    // convert to bf16
    k_f2bf<<<(SEQ * D_IN + 255) / 256, 256, 0, stream>>>(x, xb, SEQ * D_IN);
    k_f2bf<<<(G4 * D_IN + 255) / 256, 256, 0, stream>>>(W_ih, wib, G4 * D_IN);
    k_f2bf<<<(G4 * H + 255) / 256, 256, 0, stream>>>(W_hh, whhb, G4 * H);
    k_f2bf<<<(D_OUT * H + 255) / 256, 256, 0, stream>>>(W_fc, wfb, D_OUT * H);
    k_init<<<32, 256, 0, stream>>>(b_ih, b_hh, bsum, hprev, g_h, bar);

    // Phase 1: xg GEMM (262144 tiles / 8 waves)
    k_gemm_xg<<<32768, 256, 0, stream>>>(xb, wib, bsum, xg);

    // Phase 2: persistent recurrence
    k_lstm<<<NB, 256, 0, stream>>>(whhb, xg, cprev, g_h, hsb, out, bar);

    // Phase 3: FC GEMM + sigmoid (2048 tiles / 8 waves)
    k_gemm_fc<<<256, 256, 0, stream>>>(hsb, wfb, b_fc, out);
}
